// Model_42769284334148
// MI455X (gfx1250) — compile-verified, matrix operand order checked by reading, not run on hardware
//
#include <hip/hip_runtime.h>
#include <hip/hip_bf16.h>

typedef __attribute__((ext_vector_type(16))) _Float16 v16h;
typedef __attribute__((ext_vector_type(8)))  _Float16 v8h;
typedef __attribute__((ext_vector_type(8)))  float    v8f;

#define RNN_HID 64
#define SEQ_LEN 50
#define NFEAT   17
#define WAVES_PER_WG 8

// ---------------- LDS layout (dynamic shared) ----------------
// B-fragments: [frag][lane][16 f16]  -> 1024 bytes per fragment
//   frags 0..3   : W_td^T  [32(k) x 64(n)]   (k<17 valid)   ntile = F
//   frags 4..7   : W_reg^T [64(k) x 32(n)]   (n<17 valid)   F = 4 + kt*2 + ntl
//   frags 8..39  : W_ih^T  [64(k) x 256(n)]  (k<34 valid)   F = 8 + nt*2 + kt
//   frags 40..71 : W_hh^T  [64(k) x 256(n)]                 F = 40 + nt*2 + kt
#define FRAG_BYTES   1024
#define NFRAGS       72
#define OFF_FRAGS    0
#define OFF_BIAS     (NFRAGS * FRAG_BYTES)          // f32[256] = b_ih + b_hh
#define OFF_BTD      (OFF_BIAS + 1024)              // f32[64]
#define OFF_BREG     (OFF_BTD + 256)                // f32[32] (zero padded)
#define OFF_WOUT     (OFF_BREG + 128)               // f32[64]
#define OFF_WSCR     (OFF_WOUT + 256)               // per-wave scratch
#define WSCR_BYTES   4096                           // inp f16[16*64] + h f16[16*64]
#define SMEM_TOTAL   (OFF_WSCR + WAVES_PER_WG * WSCR_BYTES)

#define WMMA(a, b, c) __builtin_amdgcn_wmma_f32_16x16x32_f16(false, (a), false, (b), (short)0, (c), false, false)

__device__ __forceinline__ v8f v8f_splat(float x) {
    v8f r;
#pragma unroll
    for (int i = 0; i < 8; ++i) r[i] = x;
    return r;
}

__device__ __forceinline__ v16h cat16(v8h lo, v8h hi) {
    v16h r;
#pragma unroll
    for (int i = 0; i < 8; ++i) { r[i] = lo[i]; r[8 + i] = hi[i]; }
    return r;
}

// Load a B-fragment (32x16 f16, K-major per lane) staged in LDS.
__device__ __forceinline__ v16h load_bfrag(const char* sm, int F, int lane) {
    const char* p = sm + OFF_FRAGS + F * FRAG_BYTES + lane * 32;
    return cat16(*(const v8h*)p, *(const v8h*)(p + 16));
}

// Load an A-fragment (16x32 f16) from a row-major f16[16][64] LDS tile, k-tile kt.
// A layout: lane m (0-15) khalf 0: slots0-7 = k 0..7, slots8-15 = k 16..23 (khalf 1: +8).
__device__ __forceinline__ v16h load_afrag(const _Float16* tile, int lane, int kt) {
    int m = lane & 15, kh = lane >> 4;
    int kb = 32 * kt + 8 * kh;
    return cat16(*(const v8h*)(tile + m * 64 + kb),
                 *(const v8h*)(tile + m * 64 + kb + 16));
}

__device__ __forceinline__ float fast_sigmoid(float x) {
    return __builtin_amdgcn_rcpf(1.f + __expf(-x));
}
__device__ __forceinline__ float fast_tanh(float x) {
    return 2.f * __builtin_amdgcn_rcpf(1.f + __expf(-2.f * x)) - 1.f;
}

__global__ void __launch_bounds__(256) grud_main(
    const float* __restrict__ values, const float* __restrict__ masks,
    const float* __restrict__ deltas,
    const float* __restrict__ W_td, const float* __restrict__ b_td,
    const float* __restrict__ W_reg, const float* __restrict__ b_reg,
    const float* __restrict__ W_ih, const float* __restrict__ W_hh,
    const float* __restrict__ b_ih, const float* __restrict__ b_hh,
    const float* __restrict__ W_out, const float* __restrict__ b_out,
    float* __restrict__ out_imp, float* __restrict__ out_y,
    float* __restrict__ ws_part, int ntiles)
{
    extern __shared__ char smem[];
    const int tid  = threadIdx.x;
    const int lane = tid & 31;
    const int wave = tid >> 5;

    // ---------------- one-time weight fragment staging ----------------
    for (int p = tid; p < NFRAGS * 32; p += 256) {
        int F = p >> 5, ln = p & 31;
        int nl = ln & 15, khp = ln >> 4;
        v8h lo, hi;
#pragma unroll
        for (int s = 0; s < 16; ++s) {
            int kin = (s < 8) ? (8 * khp + s) : (16 + 8 * khp + (s - 8));
            float v = 0.f;
            if (F < 4)        { int kg = kin, ng = F * 16 + nl;
                                if (kg < NFEAT) v = W_td[ng * NFEAT + kg]; }
            else if (F < 8)   { int f = F - 4, kt = f >> 1, ntl = f & 1;
                                int kg = 32 * kt + kin, ng = 16 * ntl + nl;
                                if (ng < NFEAT) v = W_reg[ng * RNN_HID + kg]; }
            else if (F < 40)  { int f = F - 8, nt = f >> 1, kt = f & 1;
                                int kg = 32 * kt + kin, ng = 16 * nt + nl;
                                if (kg < 2 * NFEAT) v = W_ih[ng * (2 * NFEAT) + kg]; }
            else              { int f = F - 40, nt = f >> 1, kt = f & 1;
                                int kg = 32 * kt + kin, ng = 16 * nt + nl;
                                v = W_hh[ng * RNN_HID + kg]; }
            if (s < 8) lo[s] = (_Float16)v; else hi[s - 8] = (_Float16)v;
        }
        char* dst = smem + OFF_FRAGS + F * FRAG_BYTES + ln * 32;
        *(v8h*)dst = lo;
        *(v8h*)(dst + 16) = hi;
    }
    float* s_bias = (float*)(smem + OFF_BIAS);
    float* s_btd  = (float*)(smem + OFF_BTD);
    float* s_breg = (float*)(smem + OFF_BREG);
    float* s_wout = (float*)(smem + OFF_WOUT);
    if (tid < 256) s_bias[tid] = b_ih[tid] + b_hh[tid];
    if (tid < 64)  s_btd[tid]  = b_td[tid];
    if (tid < 32)  s_breg[tid] = (tid < NFEAT) ? b_reg[tid] : 0.f;
    if (tid < 64)  s_wout[tid] = W_out[tid];
    {   // zero per-wave scratch (keeps inp cols 34..63 at zero forever)
        unsigned* w0 = (unsigned*)(smem + OFF_WSCR + wave * WSCR_BYTES);
        for (int i = lane; i < WSCR_BYTES / 4; i += 32) w0[i] = 0u;
    }
    __syncthreads();

    const int tile = blockIdx.x * WAVES_PER_WG + wave;
    if (tile >= ntiles) return;

    _Float16* t_inp = (_Float16*)(smem + OFF_WSCR + wave * WSCR_BYTES); // f16[16][64]
    _Float16* t_h   = t_inp + 16 * 64;                                  // f16[16][64]

    const int m_a = lane & 15;          // A-layout row
    const int kh  = lane >> 4;          // A-layout k-half
    const int cn  = lane & 15;          // C-layout column (within n-tile)
    const int cmb = (lane >> 4) * 8;    // C-layout row base
    const size_t rowstride = (size_t)SEQ_LEN * NFEAT;

    v8f hC[4], cC[4];                   // h,c : [16 x 64] f32 in C layout
#pragma unroll
    for (int q = 0; q < 4; ++q) { hC[q] = v8f_splat(0.f); cC[q] = v8f_splat(0.f); }

    for (int t = 0; t < SEQ_LEN; ++t) {
        // ---- delta tile -> A fragment (K = 17 padded to 32) ----
        const float* drow = deltas + (size_t)(tile * 16 + m_a) * rowstride + (size_t)t * NFEAT;
        v16h dA;
#pragma unroll
        for (int s = 0; s < 8; ++s) dA[s] = (_Float16)drow[8 * kh + s];
        dA[8] = (kh == 0) ? (_Float16)drow[16] : (_Float16)0.f;
#pragma unroll
        for (int s = 9; s < 16; ++s) dA[s] = (_Float16)0.f;

        // ---- gamma = exp(-relu(d @ W_td^T + b_td)); h *= gamma ----
#pragma unroll
        for (int q = 0; q < 4; ++q) {
            v8f g = WMMA(dA, load_bfrag(smem, q, lane), v8f_splat(0.f));
            float bb = s_btd[q * 16 + cn];
#pragma unroll
            for (int j = 0; j < 8; ++j) {
                float v = g[j] + bb;
                v = v > 0.f ? v : 0.f;
                hC[q][j] *= __expf(-v);
            }
        }

        // ---- decayed h: C layout -> LDS f16 -> A fragments ----
#pragma unroll
        for (int q = 0; q < 4; ++q)
#pragma unroll
            for (int j = 0; j < 8; ++j)
                t_h[(cmb + j) * 64 + (cn + 16 * q)] = (_Float16)hC[q][j];
        __asm__ volatile("s_wait_dscnt 0" ::: "memory");
        v16h hA0 = load_afrag(t_h, lane, 0);
        v16h hA1 = load_afrag(t_h, lane, 1);

        // ---- x_h = h @ W_reg^T + b_reg  (cols 0..16 valid) ----
        v8f xh0 = v8f_splat(s_breg[cn]);
        v8f xh1 = v8f_splat(s_breg[16 + cn]);
        xh0 = WMMA(hA0, load_bfrag(smem, 4 + 0, lane), xh0);
        xh1 = WMMA(hA0, load_bfrag(smem, 4 + 1, lane), xh1);
        xh0 = WMMA(hA1, load_bfrag(smem, 4 + 2, lane), xh0);
        xh1 = WMMA(hA1, load_bfrag(smem, 4 + 3, lane), xh1);

        // ---- x, m in C layout; x_c; loss partials; imputations; inp tile ----
        float pn = 0.f, pd = 0.f;
        const size_t tbase = (size_t)t * NFEAT;
#pragma unroll
        for (int j = 0; j < 8; ++j) {
            size_t ro = (size_t)(tile * 16 + cmb + j) * rowstride + tbase;
            float xx = values[ro + cn];
            float mm = masks[ro + cn];
            xx = (xx == xx) ? xx : 0.f;                 // NaN -> 0
            float xc = mm * xx + (1.f - mm) * xh0[j];
            pn += fabsf(xx - xh0[j]) * mm;
            pd += mm;
            out_imp[ro + cn] = xc;
            t_inp[(cmb + j) * 64 + cn]       = (_Float16)xc;  // inp col n    = x_c
            t_inp[(cmb + j) * 64 + 17 + cn]  = (_Float16)mm;  // inp col 17+n = m
            if (cn == 0) {                                    // feature column 16
                float x2 = values[ro + 16];
                float m2 = masks[ro + 16];
                x2 = (x2 == x2) ? x2 : 0.f;
                float xc2 = m2 * x2 + (1.f - m2) * xh1[j];
                pn += fabsf(x2 - xh1[j]) * m2;
                pd += m2;
                out_imp[ro + 16] = xc2;
                t_inp[(cmb + j) * 64 + 16] = (_Float16)xc2;
                t_inp[(cmb + j) * 64 + 33] = (_Float16)m2;
            }
        }
#pragma unroll
        for (int off = 16; off > 0; off >>= 1) {
            pn += __shfl_xor(pn, off, 32);
            pd += __shfl_xor(pd, off, 32);
        }
        if (lane == 0) {
            ws_part[2 * ((size_t)t * ntiles + tile)]     = pn;
            ws_part[2 * ((size_t)t * ntiles + tile) + 1] = pd;
        }
        __asm__ volatile("s_wait_dscnt 0" ::: "memory");
        v16h iA0 = load_afrag(t_inp, lane, 0);
        v16h iA1 = load_afrag(t_inp, lane, 1);

        // ---- gates = inp @ W_ih^T + h @ W_hh^T + bias; LSTM update ----
#pragma unroll
        for (int ct = 0; ct < 4; ++ct) {   // hidden 16-col chunk
            v8f acc[4];
#pragma unroll
            for (int gi = 0; gi < 4; ++gi) {   // i, f, g, o
                int nt = gi * 4 + ct;
                v8f a = v8f_splat(s_bias[nt * 16 + cn]);
                a = WMMA(iA0, load_bfrag(smem, 8 + nt * 2 + 0, lane), a);
                a = WMMA(iA1, load_bfrag(smem, 8 + nt * 2 + 1, lane), a);
                a = WMMA(hA0, load_bfrag(smem, 40 + nt * 2 + 0, lane), a);
                a = WMMA(hA1, load_bfrag(smem, 40 + nt * 2 + 1, lane), a);
                acc[gi] = a;
            }
#pragma unroll
            for (int j = 0; j < 8; ++j) {
                float ig = fast_sigmoid(acc[0][j]);
                float fg = fast_sigmoid(acc[1][j]);
                float gg = fast_tanh(acc[2][j]);
                float og = fast_sigmoid(acc[3][j]);
                float cc = fg * cC[ct][j] + ig * gg;
                cC[ct][j] = cc;
                hC[ct][j] = og * fast_tanh(cc);
            }
        }
    }

    // ---- y = h @ W_out^T + b_out ----
    float py[8];
#pragma unroll
    for (int j = 0; j < 8; ++j) {
        float s = 0.f;
#pragma unroll
        for (int q = 0; q < 4; ++q) s += hC[q][j] * s_wout[q * 16 + cn];
        py[j] = s;
    }
#pragma unroll
    for (int off = 8; off > 0; off >>= 1)
#pragma unroll
        for (int j = 0; j < 8; ++j) py[j] += __shfl_xor(py[j], off, 32);
    if (cn == 0) {
        float bo = b_out[0];
#pragma unroll
        for (int j = 0; j < 8; ++j)
            out_y[tile * 16 + cmb + j] = py[j] + bo;
    }
}

// Second pass: x_loss = sum_t  num_t / (den_t + 1e-5), reducing per-tile partials.
__global__ void __launch_bounds__(256) grud_loss(const float* __restrict__ ws,
                                                 float* __restrict__ out, int ntiles)
{
    __shared__ float rn[256];
    __shared__ float rd[256];
    int tid = threadIdx.x;
    float xl = 0.f;
    for (int t = 0; t < SEQ_LEN; ++t) {
        float n = 0.f, d = 0.f;
        for (int i = tid; i < ntiles; i += 256) {
            n += ws[2 * ((size_t)t * ntiles + i)];
            d += ws[2 * ((size_t)t * ntiles + i) + 1];
        }
        rn[tid] = n; rd[tid] = d;
        __syncthreads();
        for (int s = 128; s > 0; s >>= 1) {
            if (tid < s) { rn[tid] += rn[tid + s]; rd[tid] += rd[tid + s]; }
            __syncthreads();
        }
        if (tid == 0) xl += rn[0] / (rd[0] + 1e-5f);
        __syncthreads();
    }
    if (tid == 0) out[0] = xl;
}

extern "C" void kernel_launch(void* const* d_in, const int* in_sizes, int n_in,
                              void* d_out, int out_size, void* d_ws, size_t ws_size,
                              hipStream_t stream)
{
    const float* values = (const float*)d_in[0];
    const float* masks  = (const float*)d_in[1];
    const float* deltas = (const float*)d_in[2];
    const float* W_td   = (const float*)d_in[3];
    const float* b_td   = (const float*)d_in[4];
    const float* W_reg  = (const float*)d_in[5];
    const float* b_reg  = (const float*)d_in[6];
    const float* W_ih   = (const float*)d_in[7];
    const float* W_hh   = (const float*)d_in[8];
    const float* b_ih   = (const float*)d_in[9];
    const float* b_hh   = (const float*)d_in[10];
    const float* W_out  = (const float*)d_in[11];
    const float* b_out  = (const float*)d_in[12];

    float* out = (float*)d_out;
    const int B      = in_sizes[0] / (SEQ_LEN * NFEAT);   // 8192
    const int ntiles = B / 16;                            // 512
    const int nblk   = (ntiles + WAVES_PER_WG - 1) / WAVES_PER_WG;

    float* imp = out + 1;                                  // [B,50,17]
    float* y   = out + 1 + (size_t)B * SEQ_LEN * NFEAT;    // [B,1]
    float* ws  = (float*)d_ws;                             // 2*50*ntiles f32 partials

    grud_main<<<nblk, 256, SMEM_TOTAL, stream>>>(
        values, masks, deltas, W_td, b_td, W_reg, b_reg,
        W_ih, W_hh, b_ih, b_hh, W_out, b_out,
        imp, y, ws, ntiles);
    grud_loss<<<1, 256, 0, stream>>>(ws, out, ntiles);
}
